// CorrelationModule_37495064494718
// MI455X (gfx1250) — compile-verified, matrix operand order checked by reading, not run on hardware
//
#include <hip/hip_runtime.h>
#include <hip/hip_bf16.h>
#include <cstdint>

// ---------------------------------------------------------------------------
// Correlation module (RAFT-style) for MI455X (gfx1250, wave32, WMMA).
// Conv layers = implicit GEMM on v_wmma_f32_16x16x32_f16 with 2x2 register
// tiling (2 output rows x 2 cout-tiles = 4 WMMAs per loaded A/B set),
// compile-time channel-loop unrolling, and x-edge specialization so the
// interior hot path has zero mask instructions.
// ---------------------------------------------------------------------------

typedef _Float16 v8h  __attribute__((ext_vector_type(8)));
typedef _Float16 v16h __attribute__((ext_vector_type(16)));
typedef float    v8f  __attribute__((ext_vector_type(8)));

#define HH 64
#define WW 96
#define CC 32
#define NPIX (HH*WW)        // 6144
#define NIMG 98             // B * D * D = 2*49
#define CHUNK 14            // images per chunk (98 = 7*14)

#define SHUF16(lo, hi) __builtin_shufflevector(lo, hi, 0,1,2,3,4,5,6,7,8,9,10,11,12,13,14,15)
#define WMMA(A, B, C) __builtin_amdgcn_wmma_f32_16x16x32_f16(false, A, false, B, (short)0, C, false, false)

// ---------------------------------------------------------------------------
// Weight repack: OIHW f32 (cout,cin,3,3) -> f16 blocks [K/32][cout][32],
// K index k = (ky*3+kx)*cin + ci  (matches A-fragment gather order).
// ---------------------------------------------------------------------------
__global__ void repack3x3(const float* __restrict__ src, _Float16* __restrict__ dst,
                          int cin, int cout) {
  int t = blockIdx.x * blockDim.x + threadIdx.x;
  int total = 9 * cin * cout;
  if (t >= total) return;
  int j = t & 31;
  int r = t >> 5;
  int n = r % cout;
  int blk = r / cout;
  int k = blk * 32 + j;
  int seg = k / cin;
  int ci = k - seg * cin;
  int ky = seg / 3, kx = seg - ky * 3;
  dst[t] = (_Float16)src[((size_t)n * cin + ci) * 9 + ky * 3 + kx];
}

// Deconv weights (cin=64,cout=32,kh=4,kw=4) -> per-phase f16 blocks.
__global__ void repack_deconv(const float* __restrict__ src, _Float16* __restrict__ dst) {
  int t = blockIdx.x * blockDim.x + threadIdx.x;
  if (t >= 4 * 8 * 32 * 32) return;
  int ph = t >> 13;
  int r  = t & 8191;
  int j  = r & 31;
  int r2 = r >> 5;
  int n  = r2 & 31;
  int blk = r2 >> 5;
  int k = blk * 32 + j;
  int tap = k >> 6;
  int ci = k & 63;
  int th = tap >> 1, tw = tap & 1;
  int py = ph >> 1, px = ph & 1;
  int kh = (py == 0) ? (th ? 3 : 1) : (th ? 2 : 0);
  int kw = (px == 0) ? (tw ? 3 : 1) : (tw ? 2 : 0);
  dst[t] = (_Float16)src[(((size_t)ci * 32 + n) * 4 + kh) * 4 + kw];
}

// ---------------------------------------------------------------------------
// Build matching volume (grid-sample + concat) for one chunk of images.
// Output NHWC f16: [CHUNK, 64, 96, 64]  (ch 0..31 = f1, 32..63 = sampled f2)
// ---------------------------------------------------------------------------
__global__ void build_mvol(const float* __restrict__ f1, const float* __restrict__ f2,
                           const float* __restrict__ coords, _Float16* __restrict__ mvol,
                           int lvl, int m_start, int H2, int W2) {
  int ch = threadIdx.x;
  int p = blockIdx.x * blockDim.y + threadIdx.y;
  if (p >= CHUNK * NPIX) return;
  int li = p / NPIX;
  int pix = p - li * NPIX;
  int m = m_start + li;
  int b = m / 49;
  int ij = m - b * 49;
  int di = ij / 7, dj = ij - di * 7;
  int y = pix / WW, x = pix - y * WW;

  float scale = 1.0f / (float)(1 << lvl);
  float cx = coords[((size_t)(b * 2 + 0) * HH + y) * WW + x];
  float cy = coords[((size_t)(b * 2 + 1) * HH + y) * WW + x];
  float sx = cx * scale + (float)(di - 3);
  float sy = cy * scale + (float)(dj - 3);

  float x0 = floorf(sx), y0 = floorf(sy);
  float x1 = x0 + 1.0f, y1 = y0 + 1.0f;
  float wa = (x1 - sx) * (y1 - sy);
  float wb = (x1 - sx) * (sy - y0);
  float wc = (sx - x0) * (y1 - sy);
  float wd = (sx - x0) * (sy - y0);

  const float* f2c = f2 + (size_t)(b * CC + ch) * H2 * W2;
  float wlim = (float)(W2 - 1), hlim = (float)(H2 - 1);

  auto gat = [&](float xi, float yi) -> float {
    bool valid = (xi >= 0.0f) && (xi <= wlim) && (yi >= 0.0f) && (yi <= hlim);
    float xc = fminf(fmaxf(xi, 0.0f), wlim);
    float yc = fminf(fmaxf(yi, 0.0f), hlim);
    float v = f2c[(int)yc * W2 + (int)xc];
    return valid ? v : 0.0f;
  };

  float s = gat(x0, y0) * wa + gat(x0, y1) * wb + gat(x1, y0) * wc + gat(x1, y1) * wd;

  _Float16* op = mvol + ((size_t)li * NPIX + pix) * 64;
  op[ch]      = (_Float16)f1[((size_t)(b * CC + ch) * HH + y) * WW + x];
  op[32 + ch] = (_Float16)s;
}

// ---------------------------------------------------------------------------
// One conv tap: CBLK k-blocks, 2 A rows x 2 B tiles -> 4 WMMAs per k-block.
// MASKED=false is the interior hot path (no selects at all).
// ---------------------------------------------------------------------------
template<int CBLK, bool MASKED>
__device__ __forceinline__ void conv_tap(const _Float16* __restrict__ ap0,
                                         const _Float16* __restrict__ ap1,
                                         const _Float16* __restrict__ bp, size_t bstep,
                                         bool m0, bool m1,
                                         v8f& a00, v8f& a01, v8f& a10, v8f& a11) {
  const v8h vz = {};
#pragma unroll
  for (int cb = 0; cb < CBLK; ++cb) {
    v8h a0lo = *(const v8h*)ap0;
    v8h a0hi = *(const v8h*)(ap0 + 16);
    v8h a1lo = *(const v8h*)ap1;
    v8h a1hi = *(const v8h*)(ap1 + 16);
    if (MASKED) {
      a0lo = m0 ? a0lo : vz;  a0hi = m0 ? a0hi : vz;
      a1lo = m1 ? a1lo : vz;  a1hi = m1 ? a1hi : vz;
    }
    v16h A0 = SHUF16(a0lo, a0hi);
    v16h A1 = SHUF16(a1lo, a1hi);
    v16h B0 = SHUF16(*(const v8h*)bp, *(const v8h*)(bp + 16));
    v16h B1 = SHUF16(*(const v8h*)(bp + 512), *(const v8h*)(bp + 528));
    a00 = WMMA(A0, B0, a00);
    a01 = WMMA(A0, B1, a01);
    a10 = WMMA(A1, B0, a10);
    a11 = WMMA(A1, B1, a11);
    ap0 += 32; ap1 += 32; bp += bstep;
  }
}

// ---------------------------------------------------------------------------
// 3x3 conv (pad 1), implicit GEMM on WMMA, fused BN+ReLU.
// in NHWC f16 [G,Hin,Win,cin=32*CBLK]; wpk f16 [K/32][cout][32]; out NHWC f16.
// One wave = 2 rows x 16 px x 32 cout (4 accumulators, 4 WMMAs/k-block).
// Unconditional loads rely on in-workspace guard slack; OOB contributions are
// zeroed (masked tap) or the tap row is skipped wave-uniformly.
// ---------------------------------------------------------------------------
template<int CBLK, int STRIDE>
__global__ void conv3x3_wmma(const _Float16* __restrict__ in,
                             const _Float16* __restrict__ wpk,
                             const float* __restrict__ gamma,
                             const float* __restrict__ beta,
                             _Float16* __restrict__ out,
                             int G, int Hin, int Win, int Hout, int Wout, int cout) {
  const int cin = CBLK * 32;
  const int xt = Wout >> 4;                  // column tiles per row
  const int rtiles = (Hout >> 1) * xt;       // row-pair tiles
  const int nt2 = cout >> 5;                 // cout pairs
  int wid = blockIdx.x * (blockDim.x >> 5) + (threadIdx.x >> 5);
  int total = G * rtiles * nt2;
  if (wid >= total) return;

  int img = wid / (rtiles * nt2);
  int r = wid - img * rtiles * nt2;
  int rp = r / nt2;
  int ntp = r - rp * nt2;
  int yp = rp / xt;
  int x0 = (rp - yp * xt) << 4;
  int y0 = yp << 1;

  int lane = threadIdx.x & 31;
  int col = lane & 15;
  int kh8 = (lane >> 4) << 3;
  int n0 = (ntp << 5) + col;

  bool xedge = (x0 == 0) || (STRIDE == 1 && x0 + 16 == Wout);
  int xbase = (x0 + col) * STRIDE - 1;       // xi = xbase + kx

  const size_t bstep = (size_t)cout * 32;
  const size_t rowpitch = (size_t)Win * cin;
  const _Float16* imb = in + (size_t)img * Hin * rowpitch;

  v8f a00 = {}, a01 = {}, a10 = {}, a11 = {};

#pragma clang loop unroll(disable)
  for (int ky = 0; ky < 3; ++ky) {
    int yi0 = y0 * STRIDE + ky - 1;
    int yi1 = yi0 + STRIDE;
    bool ok0 = (unsigned)yi0 < (unsigned)Hin;
    bool ok1 = (unsigned)yi1 < (unsigned)Hin;
    if (!ok0 && !ok1) continue;              // uniform: skip whole tap row
    const _Float16* rb0 = imb + (size_t)yi0 * rowpitch;
    const _Float16* rb1 = imb + (size_t)yi1 * rowpitch;
#pragma clang loop unroll(disable)
    for (int kx = 0; kx < 3; ++kx) {
      int xi = xbase + kx;
      const _Float16* ap0 = rb0 + (size_t)xi * cin + kh8;
      const _Float16* ap1 = rb1 + (size_t)xi * cin + kh8;
      const _Float16* bp = wpk + (size_t)(ky * 3 + kx) * CBLK * bstep + (size_t)n0 * 32 + kh8;
      if (xedge || !(ok0 && ok1)) {
        bool v = !xedge || ((unsigned)xi < (unsigned)Win);
        conv_tap<CBLK, true>(ap0, ap1, bp, bstep, ok0 && v, ok1 && v, a00, a01, a10, a11);
      } else {
        conv_tap<CBLK, false>(ap0, ap1, bp, bstep, true, true, a00, a01, a10, a11);
      }
    }
  }

  float g0 = gamma[n0], bt0 = beta[n0];
  float g1 = gamma[n0 + 16], bt1 = beta[n0 + 16];
  _Float16* ob0 = out + (((size_t)img * Hout + y0) * Wout + x0) * cout + n0;
  _Float16* ob1 = ob0 + (size_t)Wout * cout;
#pragma unroll
  for (int j = 0; j < 8; ++j) {
    int px = j + kh8;
    ob0[(size_t)px * cout]      = (_Float16)fmaxf(a00[j] * g0 + bt0, 0.0f);
    ob0[(size_t)px * cout + 16] = (_Float16)fmaxf(a01[j] * g1 + bt1, 0.0f);
    ob1[(size_t)px * cout]      = (_Float16)fmaxf(a10[j] * g0 + bt0, 0.0f);
    ob1[(size_t)px * cout + 16] = (_Float16)fmaxf(a11[j] * g1 + bt1, 0.0f);
  }
}

// ---------------------------------------------------------------------------
// ConvTranspose2d(64->32, k=4, s=2, p=1) as 4 phase-GEMMs (K=256), BN+ReLU.
// in NHWC f16 [G,32,48,64]; out NHWC f16 [G,64,96,32].
// ---------------------------------------------------------------------------
__global__ void deconv_wmma(const _Float16* __restrict__ in,
                            const _Float16* __restrict__ wpk,
                            const float* __restrict__ gamma,
                            const float* __restrict__ beta,
                            _Float16* __restrict__ out, int G) {
  const int mtiles = (32 * 48) >> 4;       // 96 per phase
  int wid = blockIdx.x * (blockDim.x >> 5) + (threadIdx.x >> 5);
  int total = G * 4 * mtiles;
  if (wid >= total) return;

  int img = wid / (4 * mtiles);
  int r = wid - img * 4 * mtiles;
  int ph = r / mtiles;
  int mt = r - ph * mtiles;
  int py = ph >> 1, px = ph & 1;

  int lane = threadIdx.x & 31;
  int col = lane & 15;
  int kh8 = (lane >> 4) << 3;

  int m0 = mt << 4;
  int sy = m0 / 48;                        // wave-uniform sub-row
  int sx0 = m0 - sy * 48;
  int n0 = col;

  v8f acc0 = {}, acc1 = {};
  const v8h vz = {};
  const _Float16* wph = wpk + (size_t)ph * 8192;

#pragma unroll
  for (int th = 0; th < 2; ++th) {
    int yi = sy + py - th;
    if ((unsigned)yi >= 32u) continue;
    const _Float16* rowb = in + (size_t)(img * 32 + yi) * 48 * 64;
#pragma unroll
    for (int tw = 0; tw < 2; ++tw) {
      int xi = sx0 + col + px - tw;
      bool v = (unsigned)xi < 48u;
      const _Float16* ap = rowb + (size_t)xi * 64 + kh8;
      const _Float16* bp = wph + (size_t)(th * 2 + tw) * 2048 + (size_t)n0 * 32 + kh8;
#pragma unroll
      for (int cb = 0; cb < 2; ++cb) {
        v8h alo = *(const v8h*)ap;
        v8h ahi = *(const v8h*)(ap + 16);
        alo = v ? alo : vz;
        ahi = v ? ahi : vz;
        v16h A = SHUF16(alo, ahi);
        v16h B0 = SHUF16(*(const v8h*)bp, *(const v8h*)(bp + 16));
        v16h B1 = SHUF16(*(const v8h*)(bp + 512), *(const v8h*)(bp + 528));
        acc0 = WMMA(A, B0, acc0);
        acc1 = WMMA(A, B1, acc1);
        ap += 32;
        bp += 1024;
      }
    }
  }

  float g0 = gamma[n0], bt0 = beta[n0];
  float g1 = gamma[n0 + 16], bt1 = beta[n0 + 16];
  int oy = sy * 2 + py;
  _Float16* ob = out + (((size_t)img * 64 + oy) * 96) * 32 + n0;
#pragma unroll
  for (int j = 0; j < 8; ++j) {
    int sx = sx0 + j + kh8;
    int ox = sx * 2 + px;
    ob[(size_t)ox * 32]      = (_Float16)fmaxf(acc0[j] * g0 + bt0, 0.0f);
    ob[(size_t)ox * 32 + 16] = (_Float16)fmaxf(acc1[j] * g1 + bt1, 0.0f);
  }
}

// ---------------------------------------------------------------------------
// Final 3x3 conv 32 -> 1 with bias (no ReLU). Scalar (tiny FLOPs).
// ---------------------------------------------------------------------------
__global__ void conv6_bias(const _Float16* __restrict__ in,
                           const float* __restrict__ w6,
                           const float* __restrict__ b6,
                           float* __restrict__ cost, int G, int m_start) {
  int t = blockIdx.x * blockDim.x + threadIdx.x;
  if (t >= G * NPIX) return;
  int li = t / NPIX;
  int pix = t - li * NPIX;
  int y = pix / WW, x = pix - y * WW;
  float sum = b6[0];
  for (int ky = 0; ky < 3; ++ky) {
    int yi = y + ky - 1;
    if (yi < 0 || yi >= HH) continue;
    for (int kx = 0; kx < 3; ++kx) {
      int xi = x + kx - 1;
      if (xi < 0 || xi >= WW) continue;
      const _Float16* ap = in + (((size_t)li * HH + yi) * WW + xi) * 32;
      for (int ci = 0; ci < 32; ++ci)
        sum += (float)ap[ci] * w6[ci * 9 + ky * 3 + kx];
    }
  }
  cost[(size_t)(m_start + li) * NPIX + pix] = sum;
}

// ---------------------------------------------------------------------------
// DAP: 1x1 conv over the 49 displacement channels, write final output slab.
// ---------------------------------------------------------------------------
__global__ void dap_apply(const float* __restrict__ cost,
                          const float* __restrict__ dw,
                          float* __restrict__ out, int lvl) {
  int t = blockIdx.x * blockDim.x + threadIdx.x;
  if (t >= 2 * 49 * NPIX) return;
  int b = t / (49 * NPIX);
  int r = t - b * 49 * NPIX;
  int p = r / NPIX;
  int pix = r - p * NPIX;
  const float* cb = cost + (size_t)b * 49 * NPIX + pix;
  float s = 0.0f;
  for (int q = 0; q < 49; ++q)
    s += dw[p * 49 + q] * cb[(size_t)q * NPIX];
  out[((size_t)b * 196 + lvl * 49 + p) * NPIX + pix] = s;
}

// ---------------------------------------------------------------------------
// Host orchestration.
// Input order (setup_inputs insertion order):
//   0..3  fmap1[0..3]        4..7  fmap2[0..3]       8  coords
//   9+17*lvl+k : mnet[lvl] keys {w1,g1,b1,w2,g2,b2,w3,g3,b3,w4,g4,b4,wd,gd,bd,w6,b6}
//   77..80 : dap[0..3]
// ---------------------------------------------------------------------------
extern "C" void kernel_launch(void* const* d_in, const int* in_sizes, int n_in,
                              void* d_out, int out_size, void* d_ws, size_t ws_size,
                              hipStream_t stream) {
  (void)in_sizes; (void)n_in; (void)out_size; (void)ws_size;

  const float* fmap1[4];
  const float* fmap2[4];
  for (int i = 0; i < 4; ++i) {
    fmap1[i] = (const float*)d_in[i];
    fmap2[i] = (const float*)d_in[4 + i];
  }
  const float* coords = (const float*)d_in[8];
  float* out = (float*)d_out;

  // workspace carve-up (256B aligned); bufA/bufB sit mid-workspace; the 64KB
  // tail guard absorbs the +/-1-row unconditional-load overruns of the GEMMs.
  char* ws = (char*)d_ws;
  size_t o = 0;
  auto carve = [&](size_t bytes) { size_t c = o; o += (bytes + 255) & ~(size_t)255; return c; };
  _Float16* wpk1 = (_Float16*)(ws + carve((size_t)576 * 96 * 2));
  _Float16* wpk2 = (_Float16*)(ws + carve((size_t)864 * 128 * 2));
  _Float16* wpk3 = (_Float16*)(ws + carve((size_t)1152 * 128 * 2));
  _Float16* wpk4 = (_Float16*)(ws + carve((size_t)1152 * 64 * 2));
  _Float16* wpkd = (_Float16*)(ws + carve((size_t)4 * 8192 * 2));
  float*    cost = (float*)(ws + carve((size_t)NIMG * NPIX * 4));
  _Float16* bufA = (_Float16*)(ws + carve((size_t)CHUNK * NPIX * 96 * 2));
  _Float16* bufB = (_Float16*)(ws + carve((size_t)CHUNK * NPIX * 96 * 2));
  (void)carve(65536);   // tail guard

  auto conv = [&](const _Float16* in, const _Float16* wpk, const float* g,
                  const float* b, _Float16* dst, int Hin, int Win,
                  int Hout, int Wout, int cin, int cout, int stride) {
    int waves = CHUNK * ((Hout >> 1) * (Wout >> 4)) * (cout >> 5);
    int blocks = (waves + 3) / 4;
    if (cin == 64 && stride == 1)
      conv3x3_wmma<2, 1><<<blocks, 128, 0, stream>>>(in, wpk, g, b, dst, CHUNK, Hin, Win, Hout, Wout, cout);
    else if (cin == 96 && stride == 2)
      conv3x3_wmma<3, 2><<<blocks, 128, 0, stream>>>(in, wpk, g, b, dst, CHUNK, Hin, Win, Hout, Wout, cout);
    else if (cin == 128 && stride == 1)
      conv3x3_wmma<4, 1><<<blocks, 128, 0, stream>>>(in, wpk, g, b, dst, CHUNK, Hin, Win, Hout, Wout, cout);
  };

  for (int lvl = 0; lvl < 4; ++lvl) {
    const int pb = 9 + lvl * 17;
    const float* w1 = (const float*)d_in[pb + 0];
    const float* g1 = (const float*)d_in[pb + 1];
    const float* b1 = (const float*)d_in[pb + 2];
    const float* w2 = (const float*)d_in[pb + 3];
    const float* g2 = (const float*)d_in[pb + 4];
    const float* b2 = (const float*)d_in[pb + 5];
    const float* w3 = (const float*)d_in[pb + 6];
    const float* g3 = (const float*)d_in[pb + 7];
    const float* b3 = (const float*)d_in[pb + 8];
    const float* w4 = (const float*)d_in[pb + 9];
    const float* g4 = (const float*)d_in[pb + 10];
    const float* b4 = (const float*)d_in[pb + 11];
    const float* wd = (const float*)d_in[pb + 12];
    const float* gd = (const float*)d_in[pb + 13];
    const float* bd = (const float*)d_in[pb + 14];
    const float* w6 = (const float*)d_in[pb + 15];
    const float* b6 = (const float*)d_in[pb + 16];
    const float* dapw = (const float*)d_in[77 + lvl];

    // repack weights for this level
    repack3x3<<<(576 * 96 + 255) / 256, 256, 0, stream>>>(w1, wpk1, 64, 96);
    repack3x3<<<(864 * 128 + 255) / 256, 256, 0, stream>>>(w2, wpk2, 96, 128);
    repack3x3<<<(1152 * 128 + 255) / 256, 256, 0, stream>>>(w3, wpk3, 128, 128);
    repack3x3<<<(1152 * 64 + 255) / 256, 256, 0, stream>>>(w4, wpk4, 128, 64);
    repack_deconv<<<(32768 + 255) / 256, 256, 0, stream>>>(wd, wpkd);

    int H2 = HH >> lvl, W2 = WW >> lvl;
    for (int c = 0; c < NIMG / CHUNK; ++c) {
      int m_start = c * CHUNK;
      build_mvol<<<CHUNK * NPIX / 8, dim3(32, 8), 0, stream>>>(
          fmap1[lvl], fmap2[lvl], coords, bufA, lvl, m_start, H2, W2);
      conv(bufA, wpk1, g1, b1, bufB, 64, 96, 64, 96, 64, 96, 1);
      conv(bufB, wpk2, g2, b2, bufA, 64, 96, 32, 48, 96, 128, 2);
      conv(bufA, wpk3, g3, b3, bufB, 32, 48, 32, 48, 128, 128, 1);
      conv(bufB, wpk4, g4, b4, bufA, 32, 48, 32, 48, 128, 64, 1);
      {
        int waves = CHUNK * 4 * ((32 * 48) >> 4);
        deconv_wmma<<<(waves + 3) / 4, 128, 0, stream>>>(bufA, wpkd, gd, bd, bufB, CHUNK);
      }
      conv6_bias<<<(CHUNK * NPIX + 255) / 256, 256, 0, stream>>>(
          bufB, w6, b6, cost, CHUNK, m_start);
    }
    dap_apply<<<(2 * 49 * NPIX + 255) / 256, 256, 0, stream>>>(cost, dapw, out, lvl);
  }
}
